// FirUpsample2D_16698832847478
// MI455X (gfx1250) — compile-verified, matrix operand order checked by reading, not run on hardware
//
#include <hip/hip_runtime.h>
#include <cstddef>

// ---------------- CDNA5 WMMA types ----------------
typedef float  v8f   __attribute__((ext_vector_type(8)));
typedef __bf16 bf8v  __attribute__((ext_vector_type(8)));
typedef __bf16 bf16v __attribute__((ext_vector_type(16)));

union BV { bf16v v; bf8v h[2]; };

__device__ __forceinline__ unsigned short f2bf(float f) {
    unsigned int u = __builtin_bit_cast(unsigned int, f);
    u += 0x7FFFu + ((u >> 16) & 1u);           // round-to-nearest-even
    return (unsigned short)(u >> 16);
}

__device__ __forceinline__ BV ldsBV(const char* p) {
    BV r;
    r.h[0] = *(const bf8v*)(p);
    r.h[1] = *(const bf8v*)(p + 32);
    return r;
}

#define WMMA_BF16(A, B, C) \
    __builtin_amdgcn_wmma_f32_16x16x32_bf16(false, (A).v, false, (B).v, (short)0, (C), false, false)

// LDS plan (bytes):
//   double-buffered staging: buf{0,1} = xs [19][20][32] bf16 (24320)
//                                     + wsm [9][16][32] bf16 ( 9216)  => 33536 each
//   h-tile: hs [16 oc][36][36] f32 = 82944, aliases both buffers (barrier-protected)
#define BUF_BYTES   33536
#define WS_BYTE_OFF 24320
#define SMEM_BYTES  82944

extern "C" __global__ __launch_bounds__(256)
void FirUpsample2D_wmma_kernel(const float* __restrict__ x,
                               const float* __restrict__ w,
                               const float* __restrict__ bias,
                               float* __restrict__ out) {
    extern __shared__ char smem[];
    float* hs = (float*)smem;

    const int tid   = threadIdx.x;
    const int wave  = tid >> 5;
    const int lane  = tid & 31;
    const int nlane = lane & 15;
    const int kbase = (lane >> 4) << 3;        // 0/8: per-lane-half K base

    const int grp  = wave >> 2;                // 0: q0(EE)+q3(OO), 1: q1(EO)+q2(OE)
    const int widx = wave & 3;                 // tile ownership class (t mod 4)

    const int b   = blockIdx.x;
    const int sp  = b & 15;
    const int ocg = (b >> 4) & 15;
    const int n   = b >> 8;
    const int ty0 = (sp >> 2) << 5;
    const int tx0 = (sp & 3) << 5;
    const int oc0 = ocg << 4;
    const int xrb = (ty0 >> 1) - 2;
    const int xcb = (tx0 >> 1) - 2;

    // ---- per-slot B-operand base offsets (parity-independent; hoisted out of K loop) ----
    int vboff[6];
    #pragma unroll
    for (int s = 0; s < 6; ++s) {
        int t = widx + (s << 2); if (t > 20) t = 20;   // clamp: duplicates tile 20 (benign)
        int p = (t << 4) + nlane; if (p >= 324) p = 0; // clamp pad lanes
        int u = p / 18, vv = p - u * 18;
        vboff[s] = ((u * 20 + vv) * 32 + kbase) * 2;
    }
    const int aboff = WS_BYTE_OFF + (nlane * 32 + kbase) * 2;

    // ---- staging: global f32 -> LDS bf16 (zero-fill OOB => edge padding) ----
    auto stage = [&](int icc, char* bufp) {
        const int ic0c = icc << 5;
        unsigned short* xsS = (unsigned short*)bufp;
        unsigned short* wsS = (unsigned short*)(bufp + WS_BYTE_OFF);
        for (int idx = tid; idx < 32 * 19 * 19; idx += 256) {
            int ic  = idx / 361;
            int rem = idx - ic * 361;
            int r   = rem / 19;
            int c   = rem - r * 19;
            int gr = xrb + r, gc = xcb + c;
            float v = 0.f;
            if ((unsigned)gr < 64u && (unsigned)gc < 64u)
                v = x[(((size_t)n * 256 + (ic0c + ic)) * 64 + gr) * 64 + gc];
            xsS[(r * 20 + c) * 32 + ic] = f2bf(v);
        }
        for (int idx = tid; idx < 9 * 16 * 32; idx += 256) {
            int tap  = idx % 9;
            int pair = idx / 9;
            int oc = pair >> 5, ic = pair & 31;
            float v = w[(((size_t)(oc0 + oc)) * 256 + (ic0c + ic)) * 9 + tap];
            wsS[(tap * 16 + oc) * 32 + ic] = f2bf(v);
        }
    };

    v8f accA[6] = {};   // quadrant qa (q0 or q1)
    v8f accB[6] = {};   // quadrant qb (q3 or q2)

    stage(0, smem);
    __syncthreads();

    for (int icc = 0; icc < 8; ++icc) {
        char* curp = smem + ((icc & 1) ? BUF_BYTES : 0);
        if (icc < 7)
            stage(icc + 1, smem + ((icc & 1) ? 0 : BUF_BYTES));

        if (grp == 0) {
            // taps used: q0 = {(0,0),(0,2),(2,0),(2,2)}, q3 = {(1,1)}
            BV a00 = ldsBV(curp + aboff + 0 * 1024);
            BV a02 = ldsBV(curp + aboff + 2 * 1024);
            BV a20 = ldsBV(curp + aboff + 6 * 1024);
            BV a22 = ldsBV(curp + aboff + 8 * 1024);
            BV a11 = ldsBV(curp + aboff + 4 * 1024);
            #pragma unroll
            for (int s = 0; s < 6; ++s) {
                const char* bp = curp + vboff[s];
                BV b00 = ldsBV(bp);            // (dy,dx)=(0,0)
                BV b01 = ldsBV(bp + 64);       // (0,1)
                BV b10 = ldsBV(bp + 1280);     // (1,0)
                BV b11 = ldsBV(bp + 1344);     // (1,1) shared by q0 tap(2,2) & q3 tap(1,1)
                accA[s] = WMMA_BF16(a00, b00, accA[s]);
                accA[s] = WMMA_BF16(a02, b01, accA[s]);
                accA[s] = WMMA_BF16(a20, b10, accA[s]);
                accA[s] = WMMA_BF16(a22, b11, accA[s]);
                accB[s] = WMMA_BF16(a11, b11, accB[s]);
            }
        } else {
            // taps used: q1 = {(0,1),(2,1)}, q2 = {(1,0),(1,2)}
            BV a01 = ldsBV(curp + aboff + 1 * 1024);
            BV a21 = ldsBV(curp + aboff + 7 * 1024);
            BV a10 = ldsBV(curp + aboff + 3 * 1024);
            BV a12 = ldsBV(curp + aboff + 5 * 1024);
            #pragma unroll
            for (int s = 0; s < 6; ++s) {
                const char* bp = curp + vboff[s];
                BV b01 = ldsBV(bp + 64);       // (0,1)
                BV b10 = ldsBV(bp + 1280);     // (1,0)
                BV b11 = ldsBV(bp + 1344);     // (1,1) shared by q1 tap(2,1) & q2 tap(1,2)
                accA[s] = WMMA_BF16(a01, b01, accA[s]);
                accA[s] = WMMA_BF16(a21, b11, accA[s]);
                accB[s] = WMMA_BF16(a10, b10, accB[s]);
                accB[s] = WMMA_BF16(a12, b11, accB[s]);
            }
        }
        __syncthreads();
    }

    // ---- scatter accumulators into the 36x36 h tile (aliases staging buffers) ----
    const int prA = 0,        pcA = grp ? 1 : 0;   // q0:(0,0)  q1:(0,1)
    const int prB = 1,        pcB = grp ? 0 : 1;   // q3:(1,1)  q2:(1,0)
    #pragma unroll
    for (int s = 0; s < 6; ++s) {
        int t = widx + (s << 2); if (t > 20) t = 20;
        int p = (t << 4) + nlane;
        if (p < 324) {                             // pad lanes / duplicate writes benign
            int u = p / 18, vv = p - u * 18;
            int hyA = (u << 1) + prA, hxA = (vv << 1) + pcA;
            int hyB = (u << 1) + prB, hxB = (vv << 1) + pcB;
            #pragma unroll
            for (int r = 0; r < 8; ++r) {          // D layout: M = r + 8*(lane>=16)
                hs[((kbase + r) * 36 + hyA) * 36 + hxA] = accA[s][r];
                hs[((kbase + r) * 36 + hyB) * 36 + hxB] = accB[s][r];
            }
        }
    }
    __syncthreads();

    // ---- separable 4x4 FIR ((1,3,3,1)/4 per axis) + bias, coalesced stores ----
    const int sy = tid >> 5, sx = tid & 31;
    const float kt[4] = {0.25f, 0.75f, 0.75f, 0.25f};
    for (int oc = 0; oc < 16; ++oc) {
        float bv = bias[oc0 + oc];
        #pragma unroll
        for (int rr = 0; rr < 4; ++rr) {
            int ry = sy + (rr << 3);
            float s = 0.f;
            #pragma unroll
            for (int f = 0; f < 4; ++f) {
                const float* hrow = &hs[(oc * 36 + (ry + 1 + f)) * 36 + (sx + 1)];
                float rs = kt[0] * hrow[0] + kt[1] * hrow[1]
                         + kt[2] * hrow[2] + kt[3] * hrow[3];
                s += kt[f] * rs;
            }
            out[(((size_t)n * 256 + (oc0 + oc)) * 128 + (ty0 + ry)) * 128 + (tx0 + sx)] = s + bv;
        }
    }
}

extern "C" void kernel_launch(void* const* d_in, const int* in_sizes, int n_in,
                              void* d_out, int out_size, void* d_ws, size_t ws_size,
                              hipStream_t stream) {
    (void)in_sizes; (void)n_in; (void)d_ws; (void)ws_size; (void)out_size;
    const float* x    = (const float*)d_in[0];
    const float* w    = (const float*)d_in[1];
    const float* bias = (const float*)d_in[2];
    float*       out  = (float*)d_out;

    hipFuncSetAttribute(reinterpret_cast<const void*>(FirUpsample2D_wmma_kernel),
                        hipFuncAttributeMaxDynamicSharedMemorySize, SMEM_BYTES);

    // 8 batches x 16 oc-groups x 16 spatial tiles = 2048 workgroups of 8 waves
    FirUpsample2D_wmma_kernel<<<dim3(2048), dim3(256), SMEM_BYTES, stream>>>(x, w, bias, out);
}
